// HeteroGNN_13615046328579
// MI455X (gfx1250) — matrix operand order repaired and checked by reading, not compiled, not run on hardware
//
#include <hip/hip_runtime.h>

// ---------------------------------------------------------------------------
// HeteroGNN (2-layer SAGE) for gfx1250.
//   - Dense projections via v_wmma_f32_16x16x32_bf16 (f32 accumulate)
//   - 32-row M-tile per wave: B fragments reused across 2 A fragments
//     (8 wmma : 12 b128 loads per K-step, no exec-mask games: N % 32 == 0)
//   - A operands pre-converted to bf16 row-major (no in-loop cvt)
//   - W operands pre-packed into WMMA B-fragment layout (32B/lane loads)
//   - Edge aggregation: wave-per-edge gather + f32 global atomics (L2 resident)
//   - mean @ Wl reordered to mean(x @ Wl)  -> scatter only 64 channels
//   - h_m2 (unused by reference output) is never computed
// ---------------------------------------------------------------------------

typedef __attribute__((ext_vector_type(16))) __bf16          v16bf;
typedef __attribute__((ext_vector_type(8)))  unsigned short  v8us;
typedef __attribute__((ext_vector_type(8)))  float           v8f;

__device__ __forceinline__ unsigned short f2bf(float f) {
  union { float f; unsigned u; } cv; cv.f = f;
  unsigned u = cv.u;
  u += 0x7FFFu + ((u >> 16) & 1u);        // round-to-nearest-even
  return (unsigned short)(u >> 16);
}

// f32 -> bf16 elementwise
__global__ void cvt_bf16(const float* __restrict__ in, unsigned short* __restrict__ out,
                         long long n) {
  long long t = (long long)blockIdx.x * blockDim.x + threadIdx.x;
  if (t < n) out[t] = f2bf(in[t]);
}

// Pack W[K x 64] (f32 row-major) into WMMA B-fragment layout:
//   frag[((ks*4 + nt)*32 + lane)*16 + j] = bf16( W[(ks*32 + (j>>3)*16 + (lane>>4)*8 + (j&7))*64 + nt*16 + (lane&15)] )
__global__ void prep_wfrag(const float* __restrict__ W, unsigned short* __restrict__ out, int K) {
  int t = blockIdx.x * blockDim.x + threadIdx.x;
  int total = (K / 32) * 4 * 32;
  if (t >= total) return;
  int lane = t & 31;
  int nt   = (t >> 5) & 3;
  int ks   = t >> 7;
  int half = lane >> 4;
  int col  = nt * 16 + (lane & 15);
  unsigned short* o = out + (size_t)t * 16;
#pragma unroll
  for (int j = 0; j < 16; ++j)
    o[j] = f2bf(W[(size_t)(ks * 32 + (j >> 3) * 16 + half * 8 + (j & 7)) * 64 + col]);
}

__device__ __forceinline__ v16bf load_afrag(const unsigned short* Arow, int k0, int half) {
  struct { v8us lo, hi; } af;
  af.lo = *(const v8us*)(Arow + k0 +      half * 8);
  af.hi = *(const v8us*)(Arow + k0 + 16 + half * 8);
  return __builtin_bit_cast(v16bf, af);
}

// ---------------------------------------------------------------------------
// C[N x 64] = A[N x K] @ W[K x 64]
//   A: bf16 row-major (N % 32 == 0, K % 32 == 0), Wf: pre-packed fragments.
// One wave per 32-row tile (two 16-row blocks share the B fragments);
// 8 f32 accumulators cover 32 rows x 64 columns.
// ---------------------------------------------------------------------------
__global__ void gemm64_wmma(const unsigned short* __restrict__ A,
                            const unsigned short* __restrict__ Wf,
                            float* __restrict__ C, int N, int K) {
  const int lane = threadIdx.x & 31;
  const int wave = threadIdx.x >> 5;                 // 4 waves / block
  const int tile = blockIdx.x * 4 + wave;
  const int row0 = tile * 32;
  if (row0 >= N) return;                             // wave-uniform: EXEC stays full
  const int half = lane >> 4;                        // 0 | 1
  const int mrow = lane & 15;                        // M (for A/D) or N-col (for B/D)

  v8f acc0 = {}, acc1 = {}, acc2 = {}, acc3 = {};    // rows row0   .. row0+15
  v8f acc4 = {}, acc5 = {}, acc6 = {}, acc7 = {};    // rows row0+16.. row0+31
  const unsigned short* Arow0 = A + (size_t)(row0 + mrow) * K;
  const unsigned short* Arow1 = Arow0 + (size_t)16 * K;

  for (int k0 = 0; k0 < K; k0 += 32) {
    v16bf a0 = load_afrag(Arow0, k0, half);
    v16bf a1 = load_afrag(Arow1, k0, half);
    const unsigned short* wb = Wf + ((size_t)(k0 >> 5) * 4 * 32 + lane) * 16;
    v16bf b0 = *(const v16bf*)(wb);
    v16bf b1 = *(const v16bf*)(wb + 32 * 16);
    v16bf b2 = *(const v16bf*)(wb + 64 * 16);
    v16bf b3 = *(const v16bf*)(wb + 96 * 16);
    acc0 = __builtin_amdgcn_wmma_f32_16x16x32_bf16(false, a0, false, b0, (short)0, acc0, false, false);
    acc1 = __builtin_amdgcn_wmma_f32_16x16x32_bf16(false, a0, false, b1, (short)0, acc1, false, false);
    acc2 = __builtin_amdgcn_wmma_f32_16x16x32_bf16(false, a0, false, b2, (short)0, acc2, false, false);
    acc3 = __builtin_amdgcn_wmma_f32_16x16x32_bf16(false, a0, false, b3, (short)0, acc3, false, false);
    acc4 = __builtin_amdgcn_wmma_f32_16x16x32_bf16(false, a1, false, b0, (short)0, acc4, false, false);
    acc5 = __builtin_amdgcn_wmma_f32_16x16x32_bf16(false, a1, false, b1, (short)0, acc5, false, false);
    acc6 = __builtin_amdgcn_wmma_f32_16x16x32_bf16(false, a1, false, b2, (short)0, acc6, false, false);
    acc7 = __builtin_amdgcn_wmma_f32_16x16x32_bf16(false, a1, false, b3, (short)0, acc7, false, false);
  }
  // D layout: reg r -> row = rowbase + r + 8*half, col = n0 + mrow
#pragma unroll
  for (int r = 0; r < 8; ++r) {
    float* cp = C + (size_t)(row0 + r + 8 * half) * 64 + mrow;
    cp[0]  = acc0[r];
    cp[16] = acc1[r];
    cp[32] = acc2[r];
    cp[48] = acc3[r];
  }
#pragma unroll
  for (int r = 0; r < 8; ++r) {
    float* cp = C + (size_t)(row0 + 16 + r + 8 * half) * 64 + mrow;
    cp[0]  = acc4[r];
    cp[16] = acc5[r];
    cp[32] = acc6[r];
    cp[48] = acc7[r];
  }
}

// ---------------------------------------------------------------------------
__global__ void fill0(float* __restrict__ p, long long n) {
  long long t = (long long)blockIdx.x * blockDim.x + threadIdx.x;
  if (t < n) p[t] = 0.0f;
}

__global__ void count_edges(const int* __restrict__ dst, int E, float* __restrict__ cnt) {
  int e = blockIdx.x * blockDim.x + threadIdx.x;
  if (e < E) atomicAdd(&cnt[dst[e]], 1.0f);
}

// One wave per edge: gather 64 f32 (float2/lane, 256B coalesced) + 64 f32 atomics
__global__ void scatter_add64(const float* __restrict__ P, const int* __restrict__ src,
                              const int* __restrict__ dst, int E, float* __restrict__ AGG) {
  int wid  = (int)(((long long)blockIdx.x * blockDim.x + threadIdx.x) >> 5);
  int lane = threadIdx.x & 31;
  if (wid >= E) return;
  int s = src[wid];
  int d = dst[wid];
  const float2 v = *(const float2*)(P + (size_t)s * 64 + lane * 2);
  float* ap = AGG + (size_t)d * 64 + lane * 2;
  atomicAdd(ap,     v.x);
  atomicAdd(ap + 1, v.y);
}

// In-place: QH = leaky_relu(AGG / max(cnt,1) + bl + QH), slope 0.01
__global__ void combine_lrelu(const float* __restrict__ AGG, const float* __restrict__ cnt,
                              const float* __restrict__ bl, float* __restrict__ QH,
                              long long total) {
  long long t = (long long)blockIdx.x * blockDim.x + threadIdx.x;
  if (t >= total) return;
  int i = (int)(t >> 6);
  int c = (int)(t & 63);
  float v = AGG[t] / fmaxf(cnt[i], 1.0f) + bl[c] + QH[t];
  QH[t] = (v > 0.0f) ? v : 0.01f * v;
}

// out[N x 2] = H[N x 64] @ Wo[64 x 2] + bo
__global__ void head2(const float* __restrict__ H, const float* __restrict__ Wo,
                      const float* __restrict__ bo, float* __restrict__ out, int N) {
  int i = blockIdx.x * blockDim.x + threadIdx.x;
  if (i >= N) return;
  float a0 = bo[0], a1 = bo[1];
  const float* h = H + (size_t)i * 64;
#pragma unroll 8
  for (int k = 0; k < 64; ++k) {
    float x = h[k];
    a0 += x * Wo[k * 2];
    a1 += x * Wo[k * 2 + 1];
  }
  out[(size_t)i * 2]     = a0;
  out[(size_t)i * 2 + 1] = a1;
}

// ---------------------------------------------------------------------------
extern "C" void kernel_launch(void* const* d_in, const int* in_sizes, int n_in,
                              void* d_out, int out_size, void* d_ws, size_t ws_size,
                              hipStream_t stream) {
  const float* x_trans = (const float*)d_in[0];
  const float* x_merch = (const float*)d_in[1];
  const int*   tm_src  = (const int*)d_in[2];
  const int*   tm_dst  = (const int*)d_in[3];
  const int*   mt_src  = (const int*)d_in[4];
  const int*   mt_dst  = (const int*)d_in[5];
  const float* Wl_tm1  = (const float*)d_in[6];
  const float* bl_tm1  = (const float*)d_in[7];
  const float* Wr_tm1  = (const float*)d_in[8];
  const float* Wl_mt1  = (const float*)d_in[9];
  const float* bl_mt1  = (const float*)d_in[10];
  const float* Wr_mt1  = (const float*)d_in[11];
  // d_in[12..14] = layer-2 tm weights: unused (h_m2 never feeds the output)
  const float* Wl_mt2  = (const float*)d_in[15];
  const float* bl_mt2  = (const float*)d_in[16];
  const float* Wr_mt2  = (const float*)d_in[17];
  const float* W_out   = (const float*)d_in[18];
  const float* b_out   = (const float*)d_in[19];

  const int D_T = 128, D_M = 64, H = 64;
  const int E   = in_sizes[2];
  const int N_T = in_sizes[0] / D_T;   // 200000 (multiple of 32)
  const int N_M = in_sizes[1] / D_M;   // 20000  (multiple of 32)

  // ---- workspace carve-up (bytes) ----
  char* w = (char*)d_ws;
  float* buf1  = (float*)w;               w += (size_t)N_T * H * 4;   // f32 [N_T,64]
  float* buf2  = (float*)w;               w += (size_t)N_T * H * 4;   // f32 [N_T,64]
  // xt_bf aliases buf2: same byte size (N_T*128*2 == N_T*64*4), disjoint lifetime
  unsigned short* xt_bf = (unsigned short*)buf2;                      // bf16 [N_T,128]
  float* mbuf1 = (float*)w;               w += (size_t)N_M * H * 4;   // f32 [N_M,64]
  float* mbuf2 = (float*)w;               w += (size_t)N_M * H * 4;   // f32 [N_M,64]
  float* cnt_t = (float*)w;               w += (size_t)N_T * 4;
  float* cnt_m = (float*)w;               w += (size_t)N_M * 4;
  unsigned short* xm_bf = (unsigned short*)w; w += (size_t)N_M * D_M * 2; // bf16 [N_M,64]
  unsigned short* htbf  = (unsigned short*)w; w += (size_t)N_T * H * 2;   // bf16 [N_T,64]
  unsigned short* hmbf  = (unsigned short*)w; w += (size_t)N_M * H * 2;   // bf16 [N_M,64]
  unsigned short* wf_ltm1 = (unsigned short*)w; w += (size_t)D_T * 64 * 2;
  unsigned short* wf_rtm1 = (unsigned short*)w; w += (size_t)D_M * 64 * 2;
  unsigned short* wf_lmt1 = (unsigned short*)w; w += (size_t)D_M * 64 * 2;
  unsigned short* wf_rmt1 = (unsigned short*)w; w += (size_t)D_T * 64 * 2;
  unsigned short* wf_lmt2 = (unsigned short*)w; w += (size_t)H   * 64 * 2;
  unsigned short* wf_rmt2 = (unsigned short*)w; w += (size_t)H   * 64 * 2;

  float* outp = (float*)d_out;                        // [N_T, 2]
  float* ht2  = outp + (size_t)N_T * 2;               // [N_T, 64] (second tuple element)

  const long long TtotH = (long long)N_T * H;
  const long long MtotH = (long long)N_M * H;
  const int B = 256;
  auto blks = [](long long n, int b) { return (unsigned)((n + b - 1) / b); };
  const unsigned gT  = blks(TtotH, B), gM = blks(MtotH, B);
  const unsigned gE  = blks(E, B);
  const unsigned gEw = blks((long long)E * 32, B);        // wave per edge
  const unsigned gemmT = blks(((long long)N_T + 31) / 32, 4);  // 32-row tiles, 4 waves/block
  const unsigned gemmM = blks(((long long)N_M + 31) / 32, 4);

  // ---- degree counts (shared by both layers; mt edges reused in layer 2) ----
  fill0<<<blks(N_T, B), B, 0, stream>>>(cnt_t, N_T);
  fill0<<<blks(N_M, B), B, 0, stream>>>(cnt_m, N_M);
  count_edges<<<gE, B, 0, stream>>>(tm_dst, E, cnt_m);
  count_edges<<<gE, B, 0, stream>>>(mt_dst, E, cnt_t);

  // ---- one-time operand packing ----
  cvt_bf16<<<blks((long long)N_T * D_T, B), B, 0, stream>>>(x_trans, xt_bf, (long long)N_T * D_T);
  cvt_bf16<<<blks((long long)N_M * D_M, B), B, 0, stream>>>(x_merch, xm_bf, (long long)N_M * D_M);
  prep_wfrag<<<blks(D_T * 2 * 4 * 32 / 64, B), B, 0, stream>>>(Wl_tm1, wf_ltm1, D_T);
  prep_wfrag<<<blks(D_M * 2 * 4 * 32 / 64, B), B, 0, stream>>>(Wr_tm1, wf_rtm1, D_M);
  prep_wfrag<<<blks(D_M * 2 * 4 * 32 / 64, B), B, 0, stream>>>(Wl_mt1, wf_lmt1, D_M);
  prep_wfrag<<<blks(D_T * 2 * 4 * 32 / 64, B), B, 0, stream>>>(Wr_mt1, wf_rmt1, D_T);
  prep_wfrag<<<blks(H   * 2 * 4 * 32 / 64, B), B, 0, stream>>>(Wl_mt2, wf_lmt2, H);
  prep_wfrag<<<blks(H   * 2 * 4 * 32 / 64, B), B, 0, stream>>>(Wr_mt2, wf_rmt2, H);

  // ---- layer 1, tm: h_m = lrelu(mean_tm(x_t @ Wl_tm1) + bl + x_m @ Wr_tm1) ----
  gemm64_wmma<<<gemmT, 128, 0, stream>>>(xt_bf, wf_ltm1, buf1, N_T, D_T);    // P_t
  gemm64_wmma<<<gemmM, 128, 0, stream>>>(xm_bf, wf_rtm1, mbuf1, N_M, D_M);   // Q_m
  fill0<<<gM, B, 0, stream>>>(mbuf2, MtotH);
  scatter_add64<<<gEw, B, 0, stream>>>(buf1, tm_src, tm_dst, E, mbuf2);
  combine_lrelu<<<gM, B, 0, stream>>>(mbuf2, cnt_m, bl_tm1, mbuf1, MtotH);   // H_M f32 = mbuf1
  cvt_bf16<<<gM, B, 0, stream>>>(mbuf1, hmbf, MtotH);

  // ---- layer 1, mt: h_t = lrelu(mean_mt(x_m @ Wl_mt1) + bl + x_t @ Wr_mt1) ----
  gemm64_wmma<<<gemmM, 128, 0, stream>>>(xm_bf, wf_lmt1, mbuf2, N_M, D_M);   // P_m
  gemm64_wmma<<<gemmT, 128, 0, stream>>>(xt_bf, wf_rmt1, buf1, N_T, D_T);    // Q_t (last xt_bf use)
  fill0<<<gT, B, 0, stream>>>(buf2, TtotH);                                  // xt_bf dies here
  scatter_add64<<<gEw, B, 0, stream>>>(mbuf2, mt_src, mt_dst, E, buf2);
  combine_lrelu<<<gT, B, 0, stream>>>(buf2, cnt_t, bl_mt1, buf1, TtotH);     // H_T f32 = buf1
  cvt_bf16<<<gT, B, 0, stream>>>(buf1, htbf, TtotH);

  // ---- layer 2, mt: h_t2 = lrelu(mean_mt(h_m @ Wl_mt2) + bl + h_t @ Wr_mt2) ----
  gemm64_wmma<<<gemmM, 128, 0, stream>>>(hmbf, wf_lmt2, mbuf1, N_M, H);      // P_m2
  gemm64_wmma<<<gemmT, 128, 0, stream>>>(htbf, wf_rmt2, ht2, N_T, H);        // Q_t2 -> d_out slot
  fill0<<<gT, B, 0, stream>>>(buf2, TtotH);
  scatter_add64<<<gEw, B, 0, stream>>>(mbuf1, mt_src, mt_dst, E, buf2);
  combine_lrelu<<<gT, B, 0, stream>>>(buf2, cnt_t, bl_mt2, ht2, TtotH);      // h_t2 in place

  // ---- classifier head: out = h_t2 @ W_out + b_out ----
  head2<<<blks(N_T, B), B, 0, stream>>>(ht2, W_out, b_out, outp, N_T);
}
